// GraphAttConv_62388694942248
// MI455X (gfx1250) — matrix operand order repaired and checked by reading, not compiled, not run on hardware
//
#include <hip/hip_runtime.h>

#define F_IN 512
#define HEADS 8
#define DD 64
#define LRELU_SLOPE 0.2f
#define EPS_DENOM 1e-16f

typedef __attribute__((ext_vector_type(16))) __bf16 v16bf;
typedef __attribute__((ext_vector_type(8)))  float  v8f;

// ---------------------------------------------------------------- utilities

__global__ void fill_u32_kernel(unsigned* __restrict__ p, size_t n, unsigned pat) {
    size_t i = (size_t)blockIdx.x * blockDim.x + threadIdx.x;
    size_t stride = (size_t)gridDim.x * blockDim.x;
    for (; i < n; i += stride) p[i] = pat;
}

__global__ void cvt_bf16_kernel(const float* __restrict__ in, __bf16* __restrict__ out, size_t n) {
    size_t i = (size_t)blockIdx.x * blockDim.x + threadIdx.x;
    size_t stride = (size_t)gridDim.x * blockDim.x;
    for (; i < n; i += stride) out[i] = (__bf16)in[i];
}

__device__ inline void atomicMaxF(float* addr, float v) {
    // bit-punned float max: correct for mixed signs, init bits = 0xFF800000 (-inf)
    if (v >= 0.0f) atomicMax((int*)addr, __float_as_int(v));
    else           atomicMin((unsigned int*)addr, __float_as_uint(v));
}

// ---------------------------------------------------------------- K1: dual WMMA GEMM
// One block: 256 threads (8 waves), one head, 128 node-rows, full D=64 columns.
// Computes BOTH h_e and h_r tiles (shared A fragment).
// Fragment phase: issue all 18 ds_load_b128 first, then 8 back-to-back v_wmma
// (no inter-WMMA hazards: shared A, distinct B, distinct accumulators).

#define XPAD 40   // padded row stride (bf16 elems): 80B, 16B-aligned, bank-friendly

__global__ __launch_bounds__(256) void gat_gemm_kernel(
        const __bf16* __restrict__ xb,      // [N][512] bf16
        const __bf16* __restrict__ web,     // [H][512][64] bf16
        const __bf16* __restrict__ wrb,     // [H][512][64] bf16
        float* __restrict__ he,             // [H][N][64] f32
        float* __restrict__ hr,             // [H][N][64] f32
        int n_nodes) {
    __shared__ alignas(16) __bf16 sX[128][XPAD];       // 128x32 used
    __shared__ alignas(16) __bf16 sWt[2][DD][XPAD];    // transposed: [mat][n][k], 64x32 used

    const int h    = blockIdx.y;
    const int m0   = blockIdx.x * 128;
    const int tid  = threadIdx.x;
    const int wave = tid >> 5;
    const int lane = tid & 31;
    const int lm   = lane & 15;      // row/col index within half-wave
    const int half = lane >> 4;      // 0: lanes 0-15, 1: lanes 16-31

    v8f acce[4] = {}, accr[4] = {};

    for (int k0 = 0; k0 < F_IN; k0 += 32) {
        // ---- stage 128x32 X tile: 512 uint4 chunks, 2 per thread (global_load_b128)
        #pragma unroll
        for (int i = 0; i < 2; ++i) {
            int e = tid + i * 256;       // chunk 0..511
            int r = e >> 2;              // row 0..127
            int c = (e & 3) * 8;         // bf16 col 0/8/16/24
            int g = m0 + r;
            uint4 val = {0u, 0u, 0u, 0u};
            if (g < n_nodes)
                val = *(const uint4*)(xb + (size_t)g * F_IN + k0 + c);
            *(uint4*)(&sX[r][c]) = val;
        }
        // ---- stage 32x64 W_e / W_r tiles TRANSPOSED into sWt[n][k]
        {
            int k  = tid >> 3;           // 0..31
            int nb = (tid & 7) * 8;      // 0..56
            size_t gi = ((size_t)h * F_IN + (k0 + k)) * DD + nb;
            union { uint4 u; __bf16 b[8]; } we, wr;
            we.u = *(const uint4*)(web + gi);
            wr.u = *(const uint4*)(wrb + gi);
            #pragma unroll
            for (int j = 0; j < 8; ++j) {
                sWt[0][nb + j][k] = we.b[j];
                sWt[1][nb + j][k] = wr.b[j];
            }
        }
        // prefetch next K-slice of this wave's X rows into cache
        if (k0 + 32 < F_IN) {
            int g = m0 + wave * 16 + lm;
            if (g < n_nodes)
                __builtin_prefetch(xb + (size_t)g * F_IN + k0 + 32, 0, 1);
        }
        __syncthreads();

        // ---- A fragment: two aligned 16B chunks of the row (ds_load_b128 x2)
        // half=0 -> bytes [0,16)+[16,32); half=1 -> bytes [32,48)+[48,64)
        union { v16bf v; uint4 u[2]; } af;
        {
            const uint4* xrow = (const uint4*)(&sX[wave * 16 + lm][0]);
            af.u[0] = xrow[2 * half];
            af.u[1] = xrow[2 * half + 1];
        }

        // ---- load ALL 8 B fragments first (16 x ds_load_b128)...
        union { v16bf v; uint4 u[2]; } bfr[8];
        #pragma unroll
        for (int t = 0; t < 4; ++t) {
            const uint4* bep = (const uint4*)(&sWt[0][t * 16 + lm][16 * half]);
            const uint4* brp = (const uint4*)(&sWt[1][t * 16 + lm][16 * half]);
            bfr[t].u[0]     = bep[0]; bfr[t].u[1]     = bep[1];
            bfr[4 + t].u[0] = brp[0]; bfr[4 + t].u[1] = brp[1];
        }
        // ---- ...then fire 8 WMMAs back-to-back (no inter-instruction hazards)
        #pragma unroll
        for (int t = 0; t < 4; ++t) {
            acce[t] = __builtin_amdgcn_wmma_f32_16x16x32_bf16(
                false, af.v, false, bfr[t].v, (short)0, acce[t], false, false);
            accr[t] = __builtin_amdgcn_wmma_f32_16x16x32_bf16(
                false, af.v, false, bfr[4 + t].v, (short)0, accr[t], false, false);
        }
        __syncthreads();
    }

    // C/D layout: VGPR v -> row (v + 8*half), col = lane&15 (16-lane coalesced stores)
    #pragma unroll
    for (int t = 0; t < 4; ++t) {
        #pragma unroll
        for (int v = 0; v < 8; ++v) {
            int row = m0 + wave * 16 + v + 8 * half;
            if (row < n_nodes) {
                int col = t * 16 + lm;
                size_t idx = ((size_t)h * n_nodes + row) * DD + col;
                he[idx] = acce[t][v];
                hr[idx] = accr[t][v];
            }
        }
    }
}

// ---------------------------------------------------------------- K2: per-node scores

__global__ void gat_node_scores_kernel(const float* __restrict__ he,
                                       const float* __restrict__ a,
                                       float* __restrict__ s1,
                                       float* __restrict__ s2,
                                       int n_nodes) {
    int i = blockIdx.x * blockDim.x + threadIdx.x;          // i = h*N + n
    if (i >= HEADS * n_nodes) return;
    int h = i / n_nodes;
    const float4* hv = (const float4*)(he + (size_t)i * DD);
    const float4* as = (const float4*)(a + h * 5 * DD);
    const float4* ad = (const float4*)(a + h * 5 * DD + DD);
    float acc1 = 0.0f, acc2 = 0.0f;
    #pragma unroll
    for (int q = 0; q < 16; ++q) {
        float4 v = hv[q], p = as[q], r = ad[q];
        acc1 += v.x * p.x + v.y * p.y + v.z * p.z + v.w * p.w;
        acc2 += v.x * r.x + v.y * r.y + v.z * r.z + v.w * r.w;
    }
    s1[i] = acc1;
    s2[i] = acc2;
}

// ---------------------------------------------------------------- K3: edge score + leaky relu + segment max

__global__ void gat_edge_score_kernel(const float* __restrict__ hr,
                                      const float* __restrict__ a,
                                      const float* __restrict__ s1,
                                      const float* __restrict__ s2,
                                      const int* __restrict__ ei,
                                      float* __restrict__ alphaBuf,
                                      float* __restrict__ mbuf,
                                      int n_nodes, int n_edges) {
    size_t idx = (size_t)blockIdx.x * blockDim.x + threadIdx.x;
    if (idx >= (size_t)n_edges * HEADS) return;
    int h = (int)(idx & 7);
    int e = (int)(idx >> 3);
    int src = ei[e];
    int dst = ei[(size_t)n_edges + e];

    const float4* hs  = (const float4*)(hr + ((size_t)h * n_nodes + src) * DD);
    const float4* hd  = (const float4*)(hr + ((size_t)h * n_nodes + dst) * DD);
    const float4* adf = (const float4*)(a + h * 5 * DD + 2 * DD);
    const float4* aab = (const float4*)(a + h * 5 * DD + 3 * DD);
    const float4* apr = (const float4*)(a + h * 5 * DD + 4 * DD);

    float acc = 0.0f;
    #pragma unroll 4
    for (int q = 0; q < 16; ++q) {
        float4 vs = hs[q], vd = hd[q], f = adf[q], b = aab[q], p = apr[q];
        float dx = vd.x - vs.x, dy = vd.y - vs.y, dz = vd.z - vs.z, dw = vd.w - vs.w;
        acc += f.x * dx + b.x * fabsf(dx) + p.x * (vs.x * vd.x);
        acc += f.y * dy + b.y * fabsf(dy) + p.y * (vs.y * vd.y);
        acc += f.z * dz + b.z * fabsf(dz) + p.z * (vs.z * vd.z);
        acc += f.w * dw + b.w * fabsf(dw) + p.w * (vs.w * vd.w);
    }
    float score = s1[(size_t)h * n_nodes + src] + s2[(size_t)h * n_nodes + dst] + acc;
    float al = (score > 0.0f) ? score : LRELU_SLOPE * score;
    alphaBuf[idx] = al;
    atomicMaxF(&mbuf[(size_t)h * n_nodes + src], al);
}

// ---------------------------------------------------------------- K4: exp + segment sum

__global__ void gat_edge_exp_kernel(const float* __restrict__ mbuf,
                                    const int* __restrict__ ei,
                                    float* __restrict__ alphaBuf,   // in: alpha, out: ex
                                    float* __restrict__ denom,
                                    int n_nodes, int n_edges) {
    size_t idx = (size_t)blockIdx.x * blockDim.x + threadIdx.x;
    if (idx >= (size_t)n_edges * HEADS) return;
    int h = (int)(idx & 7);
    int e = (int)(idx >> 3);
    int src = ei[e];
    float ex = __expf(alphaBuf[idx] - mbuf[(size_t)h * n_nodes + src]);
    alphaBuf[idx] = ex;
    atomicAdd(&denom[(size_t)h * n_nodes + src], ex);
}

// ---------------------------------------------------------------- K5: normalize + scatter-add messages

__global__ void gat_edge_agg_kernel(const float* __restrict__ he,
                                    const float* __restrict__ denom,
                                    const int* __restrict__ ei,
                                    const float* __restrict__ exBuf,
                                    float* __restrict__ out,        // [N][HEADS*DD]
                                    int n_nodes, int n_edges) {
    size_t idx = (size_t)blockIdx.x * blockDim.x + threadIdx.x;
    if (idx >= (size_t)n_edges * HEADS) return;
    int h = (int)(idx & 7);
    int e = (int)(idx >> 3);
    int src = ei[e];
    int dst = ei[(size_t)n_edges + e];
    float att = exBuf[idx] / (denom[(size_t)h * n_nodes + src] + EPS_DENOM);
    const float4* hv = (const float4*)(he + ((size_t)h * n_nodes + dst) * DD);
    float* op = out + (size_t)src * (HEADS * DD) + h * DD;
    #pragma unroll 4
    for (int q = 0; q < 16; ++q) {
        float4 v = hv[q];
        atomicAdd(op + 4 * q + 0, att * v.x);
        atomicAdd(op + 4 * q + 1, att * v.y);
        atomicAdd(op + 4 * q + 2, att * v.z);
        atomicAdd(op + 4 * q + 3, att * v.w);
    }
}

// ---------------------------------------------------------------- launch

extern "C" void kernel_launch(void* const* d_in, const int* in_sizes, int n_in,
                              void* d_out, int out_size, void* d_ws, size_t ws_size,
                              hipStream_t stream) {
    const float* x  = (const float*)d_in[0];
    const int*   ei = (const int*)d_in[1];
    const float* We = (const float*)d_in[2];
    const float* Wr = (const float*)d_in[3];
    const float* a  = (const float*)d_in[4];
    float* out = (float*)d_out;

    const int n_nodes = in_sizes[0] / F_IN;
    const int n_edges = in_sizes[1] / 2;

    char* ws = (char*)d_ws;
    size_t off = 0;
    auto carve = [&](size_t bytes) -> void* {
        void* p = ws + off;
        off += (bytes + 255) & ~(size_t)255;
        return p;
    };

    __bf16* xb   = (__bf16*)carve((size_t)n_nodes * F_IN * sizeof(__bf16));
    __bf16* web  = (__bf16*)carve((size_t)HEADS * F_IN * DD * sizeof(__bf16));
    __bf16* wrb  = (__bf16*)carve((size_t)HEADS * F_IN * DD * sizeof(__bf16));
    float*  he   = (float*)carve((size_t)HEADS * n_nodes * DD * sizeof(float));
    float*  hr   = (float*)carve((size_t)HEADS * n_nodes * DD * sizeof(float));
    float*  s1   = (float*)carve((size_t)HEADS * n_nodes * sizeof(float));
    float*  s2   = (float*)carve((size_t)HEADS * n_nodes * sizeof(float));
    float*  mbuf = (float*)carve((size_t)HEADS * n_nodes * sizeof(float));
    float*  dnm  = (float*)carve((size_t)HEADS * n_nodes * sizeof(float));
    float*  exb  = (float*)carve((size_t)n_edges * HEADS * sizeof(float));

    // K0: conversions + initialization
    cvt_bf16_kernel<<<4096, 256, 0, stream>>>(x, xb, (size_t)n_nodes * F_IN);
    cvt_bf16_kernel<<<1024, 256, 0, stream>>>(We, web, (size_t)HEADS * F_IN * DD);
    cvt_bf16_kernel<<<1024, 256, 0, stream>>>(Wr, wrb, (size_t)HEADS * F_IN * DD);
    fill_u32_kernel<<<1024, 256, 0, stream>>>((unsigned*)mbuf, (size_t)HEADS * n_nodes, 0xFF800000u); // -inf
    fill_u32_kernel<<<1024, 256, 0, stream>>>((unsigned*)dnm, (size_t)HEADS * n_nodes, 0u);
    fill_u32_kernel<<<4096, 256, 0, stream>>>((unsigned*)out, (size_t)out_size, 0u);

    // K1: dual bf16 WMMA GEMM -> h_e, h_r
    dim3 ggrid((n_nodes + 127) / 128, HEADS, 1);
    gat_gemm_kernel<<<ggrid, 256, 0, stream>>>(xb, web, wrb, he, hr, n_nodes);

    // K2: per-node s_src / s_dst
    int nh = HEADS * n_nodes;
    gat_node_scores_kernel<<<(nh + 255) / 256, 256, 0, stream>>>(he, a, s1, s2, n_nodes);

    // K3..K5: edge phase
    size_t neh = (size_t)n_edges * HEADS;
    int eblocks = (int)((neh + 255) / 256);
    gat_edge_score_kernel<<<eblocks, 256, 0, stream>>>(hr, a, s1, s2, ei, exb, mbuf, n_nodes, n_edges);
    gat_edge_exp_kernel<<<eblocks, 256, 0, stream>>>(mbuf, ei, exb, dnm, n_nodes, n_edges);
    gat_edge_agg_kernel<<<eblocks, 256, 0, stream>>>(he, dnm, ei, exb, out, n_nodes, n_edges);
}
